// CausalSelfAttention_31894427140364
// MI455X (gfx1250) — compile-verified, matrix operand order checked by reading, not run on hardware
//
#include <hip/hip_runtime.h>
#include <hip/hip_bf16.h>

typedef __attribute__((ext_vector_type(16))) __bf16 bf16x16;
typedef __attribute__((ext_vector_type(8)))  __bf16 bf16x8;
typedef __attribute__((ext_vector_type(8)))  float  f32x8;

#define Bsz    2
#define Tseq   2048
#define Dmodel 1024
#define Hn     16
#define HDim   64
#define Mrows  (Bsz * Tseq)   // 4096

// exp(0.125 * x) == exp2(x * 0.125 * log2(e)); v_exp_f32 is native exp2.
#define SCALE_LOG2E 0.1803368801111204f

// Native f32 -> bf16 convert (RNE); lowers to v_cvt_pk_bf16_f32 on gfx1250.
__device__ __forceinline__ __bf16 f2bf(float f) { return (__bf16)f; }

// Load a 16x32 bf16 A-style fragment (also used for B fragments of the
// transposed operand). Per ISA 7.12.2: lane m = lane&15 owns row m; half
// (lane>>4) selects K offset; elements e<8 -> K=8h+e, e>=8 -> K=8h+e+8.
__device__ __forceinline__ bf16x16 load_frag(const __bf16* base, int stride, int lane) {
  const int m = lane & 15;
  const int h = (lane >> 4) & 1;
  const __bf16* p = base + m * stride + 8 * h;
  bf16x8 lo = *(const bf16x8*)(p);
  bf16x8 hi = *(const bf16x8*)(p + 16);
  return __builtin_shufflevector(lo, hi, 0,1,2,3,4,5,6,7,8,9,10,11,12,13,14,15);
}

__device__ __forceinline__ f32x8 wmma_bf16(bf16x16 a, bf16x16 b, f32x8 c) {
  return __builtin_amdgcn_wmma_f32_16x16x32_bf16(false, a, false, b, (short)0, c, false, false);
}

// C[M,N] = X[M,K] @ W[N,K]^T + bias, M=4096, N=K=1024.
// Tile: 128(M) x 64(N), K-step 32, double-buffered LDS software pipeline.
// MODE 0: store bf16 to [b,h,t,hd] (Q or K)
// MODE 1: store bf16 to [b,h,hd,t] (V transposed)
// MODE 2: X is bf16 (ctx); store f32 row-major [M,N] (final output)
template <int MODE>
__global__ __launch_bounds__(256) void proj_gemm(
    const float* __restrict__ Xf, const __bf16* __restrict__ Xbf,
    const float* __restrict__ W, const float* __restrict__ bias,
    __bf16* __restrict__ out_bf, float* __restrict__ out_f)
{
  __shared__ __bf16 Xs[2][128 * 32];   // 16 KB
  __shared__ __bf16 Ws[2][64 * 32];    //  8 KB

  const int tid  = threadIdx.x;
  const int lane = tid & 31;
  const int wave = tid >> 5;        // 0..7 -> row strip 16*wave
  const int mTile = blockIdx.y * 128;
  const int nTile = blockIdx.x * 64;

  const int xrow = tid >> 1;          // 0..127 (16 elems/thread)
  const int xcol = (tid & 1) * 16;    // 0,16
  const int wrow = tid >> 2;          // 0..63  (8 elems/thread)
  const int wcol = (tid & 3) * 8;     // 0,8,16,24

  float  xstage[16];
  bf16x8 xstageb[2];
  float  wstage[8];

  auto load_stage = [&](int k0) {
    if constexpr (MODE == 2) {
      const __bf16* s = Xbf + (size_t)(mTile + xrow) * Dmodel + k0 + xcol;
      xstageb[0] = *(const bf16x8*)(s);
      xstageb[1] = *(const bf16x8*)(s + 8);
      __builtin_prefetch(s + 32, 0, 0);
    } else {
      const float* s = Xf + (size_t)(mTile + xrow) * Dmodel + k0 + xcol;
      #pragma unroll
      for (int j = 0; j < 16; ++j) xstage[j] = s[j];
      __builtin_prefetch(s + 32, 0, 0);
    }
    const float* s = W + (size_t)(nTile + wrow) * Dmodel + k0 + wcol;
    #pragma unroll
    for (int j = 0; j < 8; ++j) wstage[j] = s[j];
    __builtin_prefetch(s + 32, 0, 0);
  };

  auto store_stage = [&](int buf) {
    if constexpr (MODE == 2) {
      *(bf16x8*)(&Xs[buf][xrow * 32 + xcol])     = xstageb[0];
      *(bf16x8*)(&Xs[buf][xrow * 32 + xcol + 8]) = xstageb[1];
    } else {
      #pragma unroll
      for (int j = 0; j < 16; ++j) Xs[buf][xrow * 32 + xcol + j] = f2bf(xstage[j]);
    }
    #pragma unroll
    for (int j = 0; j < 8; ++j) Ws[buf][wrow * 32 + wcol + j] = f2bf(wstage[j]);
  };

  f32x8 acc0 = {}, acc1 = {}, acc2 = {}, acc3 = {};

  // Prologue: stage and commit tile 0.
  load_stage(0);
  store_stage(0);

  const int nK = Dmodel / 32;   // 32 K-steps
  for (int ks = 0; ks < nK; ++ks) {
    const int cur = ks & 1;
    __syncthreads();            // tile `cur` committed; prior reads of `cur^1` done

    if (ks + 1 < nK) load_stage((ks + 1) * 32);   // global loads overlap WMMAs

    bf16x16 a  = load_frag(&Xs[cur][(16 * wave) * 32], 32, lane);
    bf16x16 b0 = load_frag(&Ws[cur][ 0 * 32], 32, lane);
    bf16x16 b1 = load_frag(&Ws[cur][16 * 32], 32, lane);
    bf16x16 b2 = load_frag(&Ws[cur][32 * 32], 32, lane);
    bf16x16 b3 = load_frag(&Ws[cur][48 * 32], 32, lane);
    acc0 = wmma_bf16(a, b0, acc0);
    acc1 = wmma_bf16(a, b1, acc1);
    acc2 = wmma_bf16(a, b2, acc2);
    acc3 = wmma_bf16(a, b3, acc3);

    if (ks + 1 < nK) store_stage(cur ^ 1);
  }

  const int half = lane >> 4;
  const int nloc = lane & 15;
  #pragma unroll
  for (int s = 0; s < 4; ++s) {
    f32x8 acc = (s == 0) ? acc0 : (s == 1) ? acc1 : (s == 2) ? acc2 : acc3;
    #pragma unroll
    for (int r = 0; r < 8; ++r) {
      const int row = mTile + 16 * wave + r + 8 * half;   // m = r + 8*half
      const int col = nTile + 16 * s + nloc;              // n = lane&15
      const float v = acc[r] + bias[col];
      const int b  = row >> 11;     // / Tseq
      const int t  = row & 2047;
      const int h  = col >> 6;      // / HDim
      const int hd = col & 63;
      if constexpr (MODE == 0) {
        out_bf[(((size_t)(b * Hn + h)) * Tseq + t) * HDim + hd] = f2bf(v);
      } else if constexpr (MODE == 1) {
        out_bf[(((size_t)(b * Hn + h)) * HDim + hd) * Tseq + t] = f2bf(v);
      } else {
        out_f[(size_t)row * Dmodel + col] = v;
      }
    }
  }
}

// Flash-attention style: one wave owns 16 query rows; key blocks of 64.
// Causal split: every block before the diagonal block is fully unmasked
// (qRow0 % 16 == 0 and 64 | kbL imply kbL <= qRow0), so only the final
// block evaluates the mask.
// grid = (Tseq/128, Bsz*Hn), block = 256 (8 waves).
__global__ __launch_bounds__(256) void attn_kernel(
    const __bf16* __restrict__ qg, const __bf16* __restrict__ kg,
    const __bf16* __restrict__ vt, __bf16* __restrict__ ctx)
{
  __shared__ __bf16 Pl[8][16 * 64];   // per-wave P tile, row-major

  const int tid  = threadIdx.x;
  const int lane = tid & 31;
  const int wave = tid >> 5;
  const int bh = blockIdx.y;   // b*Hn + h
  const int b  = bh >> 4;
  const int h  = bh & 15;
  const int qRow0 = blockIdx.x * 128 + wave * 16;
  const int half = lane >> 4;
  const int nloc = lane & 15;

  const __bf16* qb = qg + (size_t)bh * Tseq * HDim;
  const __bf16* kb = kg + (size_t)bh * Tseq * HDim;
  const __bf16* vb = vt + (size_t)bh * HDim * Tseq;

  // Q fragments for this wave's 16 rows (K = HD = 64 -> two 32-chunks)
  bf16x16 qa0 = load_frag(qb + (size_t)qRow0 * HDim,      HDim, lane);
  bf16x16 qa1 = load_frag(qb + (size_t)qRow0 * HDim + 32, HDim, lane);

  f32x8 o0 = {}, o1 = {}, o2 = {}, o3 = {};
  float rmax[8], rsum[8];
  #pragma unroll
  for (int r = 0; r < 8; ++r) { rmax[r] = -1e30f; rsum[r] = 0.f; }

  // Process one 64-key block. `masked` folds at compile time after inlining.
  auto process_block = [&](int kb0, bool masked, bool prefetch) {
    if (prefetch) {
      __builtin_prefetch(kb + (size_t)(kb0 + 64 + lane * 2) * HDim, 0, 0);
      __builtin_prefetch(vb + (size_t)(lane * 2) * Tseq + kb0 + 64, 0, 0);
    }

    // scores: Q @ K^T for 4 key subtiles (unscaled; scale folded into exp2)
    float sc[4][8];
    #pragma unroll
    for (int sub = 0; sub < 4; ++sub) {
      const int keyCol = kb0 + 16 * sub;
      bf16x16 kf0 = load_frag(kb + (size_t)keyCol * HDim,      HDim, lane);
      bf16x16 kf1 = load_frag(kb + (size_t)keyCol * HDim + 32, HDim, lane);
      f32x8 s = {};
      s = wmma_bf16(qa0, kf0, s);
      s = wmma_bf16(qa1, kf1, s);
      #pragma unroll
      for (int r = 0; r < 8; ++r) {
        float v = s[r];
        if (masked) {
          const int key = keyCol + nloc;
          const int qr  = qRow0 + r + 8 * half;
          v = (key > qr) ? -1e30f : v;     // causal mask (diagonal block only)
        }
        sc[sub][r] = v;
      }
    }

    // hoist V fragments: global loads overlap the softmax VALU work below
    bf16x16 vf[8];
    #pragma unroll
    for (int sub = 0; sub < 4; ++sub) {
      vf[2 * sub]     = load_frag(vb + (size_t)(16 * sub) * Tseq + kb0,      Tseq, lane);
      vf[2 * sub + 1] = load_frag(vb + (size_t)(16 * sub) * Tseq + kb0 + 32, Tseq, lane);
    }

    // online softmax on unscaled scores: p = exp2((s - m) * 0.125*log2e)
    float nm[8], fac[8];
    #pragma unroll
    for (int r = 0; r < 8; ++r) {
      float m = fmaxf(fmaxf(sc[0][r], sc[1][r]), fmaxf(sc[2][r], sc[3][r]));
      for (int d = 1; d < 16; d <<= 1) m = fmaxf(m, __shfl_xor(m, d, 32));
      nm[r]  = fmaxf(rmax[r], m);
      fac[r] = exp2f((rmax[r] - nm[r]) * SCALE_LOG2E);
      rmax[r] = nm[r];
    }
    #pragma unroll
    for (int r = 0; r < 8; ++r) {
      float psum = 0.f;
      #pragma unroll
      for (int sub = 0; sub < 4; ++sub) {
        const float p = exp2f((sc[sub][r] - nm[r]) * SCALE_LOG2E);
        psum += p;
        Pl[wave][(r + 8 * half) * 64 + 16 * sub + nloc] = f2bf(p);
      }
      for (int d = 1; d < 16; d <<= 1) psum += __shfl_xor(psum, d, 32);
      rsum[r] = rsum[r] * fac[r] + psum;
    }
    #pragma unroll
    for (int r = 0; r < 8; ++r) {
      o0[r] *= fac[r]; o1[r] *= fac[r]; o2[r] *= fac[r]; o3[r] *= fac[r];
    }

    // wave-local LDS round-trip: D-layout -> A-fragment layout
    asm volatile("s_wait_dscnt 0x0" ::: "memory");
    bf16x16 pa0 = load_frag(&Pl[wave][0],  64, lane);
    bf16x16 pa1 = load_frag(&Pl[wave][32], 64, lane);

    #pragma unroll
    for (int sub = 0; sub < 4; ++sub) {
      f32x8& o = (sub == 0) ? o0 : (sub == 1) ? o1 : (sub == 2) ? o2 : o3;
      o = wmma_bf16(pa0, vf[2 * sub],     o);
      o = wmma_bf16(pa1, vf[2 * sub + 1], o);
    }
  };

  const int qMax = qRow0 + 15;
  const int kbL  = (qMax >> 6) << 6;   // diagonal block start; kbL <= qRow0
  for (int kb0 = 0; kb0 < kbL; kb0 += 64)
    process_block(kb0, false, true);   // unmasked fast path
  process_block(kbL, true, false);     // diagonal block with causal mask

  // normalize and store context as bf16 row-major [B*T, D]
  #pragma unroll
  for (int r = 0; r < 8; ++r) {
    const float inv = 1.0f / rsum[r];
    const int t = qRow0 + r + 8 * half;
    const size_t rowbase = ((size_t)(b * Tseq + t)) * Dmodel + h * HDim;
    ctx[rowbase +  0 + nloc] = f2bf(o0[r] * inv);
    ctx[rowbase + 16 + nloc] = f2bf(o1[r] * inv);
    ctx[rowbase + 32 + nloc] = f2bf(o2[r] * inv);
    ctx[rowbase + 48 + nloc] = f2bf(o3[r] * inv);
  }
}

extern "C" void kernel_launch(void* const* d_in, const int* in_sizes, int n_in,
                              void* d_out, int out_size, void* d_ws, size_t ws_size,
                              hipStream_t stream)
{
  const float* x  = (const float*)d_in[0];
  const float* Wq = (const float*)d_in[1];
  const float* bq = (const float*)d_in[2];
  const float* Wk = (const float*)d_in[3];
  const float* bk = (const float*)d_in[4];
  const float* Wv = (const float*)d_in[5];
  const float* bv = (const float*)d_in[6];
  const float* Wp = (const float*)d_in[7];
  const float* bp = (const float*)d_in[8];
  float* out = (float*)d_out;

  char* ws = (char*)d_ws;
  const size_t elems = (size_t)Mrows * Dmodel;          // 4M elems
  __bf16* qbf = (__bf16*)(ws);                          // 8 MB
  __bf16* kbf = (__bf16*)(ws + 2 * elems);              // 8 MB
  __bf16* vtb = (__bf16*)(ws + 4 * elems);              // 8 MB
  __bf16* ctx = (__bf16*)(ws + 6 * elems);              // 8 MB

  dim3 gg(Dmodel / 64, Mrows / 128);                    // (16, 32)
  proj_gemm<0><<<gg, 256, 0, stream>>>(x, nullptr, Wq, bq, qbf, nullptr);
  proj_gemm<0><<<gg, 256, 0, stream>>>(x, nullptr, Wk, bk, kbf, nullptr);
  proj_gemm<1><<<gg, 256, 0, stream>>>(x, nullptr, Wv, bv, vtb, nullptr);

  dim3 ga(Tseq / 128, Bsz * Hn);                        // (16, 32)
  attn_kernel<<<ga, 256, 0, stream>>>(qbf, kbf, vtb, ctx);

  proj_gemm<2><<<gg, 256, 0, stream>>>(nullptr, ctx, Wp, bp, nullptr, out);
}
